// Fake_SD_MHA_Model_66949950210785
// MI455X (gfx1250) — matrix operand order repaired and checked by reading, not compiled, not run on hardware
//
#include <hip/hip_runtime.h>
#include <math.h>

typedef __attribute__((ext_vector_type(16))) _Float16 v16h;
typedef __attribute__((ext_vector_type(8)))  _Float16 v8h;
typedef __attribute__((ext_vector_type(8)))  float    v8f;

#define S_LEN    2048
#define DIM      64
#define BM       128    // query rows per workgroup (8 waves x 16 queries)
#define BN       64     // keys per KV tile
#define NWAVE    8
#define NTHREADS 256

// ---------------------------------------------------------------------------
// Pre-pass: K (f32 [key][d]) -> Kh (f16 [key][d]),  V (f32 [key][d]) -> Vt (f16 [d][key])
// One 64-key block per workgroup; transpose through LDS; all I/O b128-coalesced.
// ---------------------------------------------------------------------------
__global__ __launch_bounds__(NTHREADS)
void preconvert_kernel(const float* __restrict__ K, const float* __restrict__ V,
                       _Float16* __restrict__ Kh, _Float16* __restrict__ Vt)
{
    __shared__ _Float16 tK[BN * DIM];
    __shared__ _Float16 tV[DIM * BN];

    const int tid = threadIdx.x;
    const int n0  = blockIdx.x * BN;
    const int bh  = blockIdx.y;

    const float* Kp = K + (size_t)bh * S_LEN * DIM;
    const float* Vp = V + (size_t)bh * S_LEN * DIM;

#pragma unroll
    for (int i = 0; i < 4; ++i) {
        const int id4 = tid + i * NTHREADS;          // 1024 float4 units
        const int key = id4 >> 4;
        const int d4  = (id4 & 15) << 2;
        const float4 kf = *(const float4*)(Kp + (size_t)(n0 + key) * DIM + d4);
        _Float16* kd = &tK[key * DIM + d4];
        kd[0] = (_Float16)kf.x; kd[1] = (_Float16)kf.y;
        kd[2] = (_Float16)kf.z; kd[3] = (_Float16)kf.w;
        const float4 vf = *(const float4*)(Vp + (size_t)(n0 + key) * DIM + d4);
        tV[(d4 + 0) * BN + key] = (_Float16)vf.x;
        tV[(d4 + 1) * BN + key] = (_Float16)vf.y;
        tV[(d4 + 2) * BN + key] = (_Float16)vf.z;
        tV[(d4 + 3) * BN + key] = (_Float16)vf.w;
    }
    __syncthreads();

    _Float16* Khp = Kh + (size_t)bh * S_LEN * DIM + (size_t)n0 * DIM;  // contiguous tile
    _Float16* Vtp = Vt + (size_t)bh * DIM * S_LEN;                     // row stride S_LEN
#pragma unroll
    for (int i = 0; i < 2; ++i) {
        const int idx = tid + i * NTHREADS;          // 512 units of 8 halves
        *(v8h*)(Khp + (size_t)idx * 8) = *(const v8h*)&tK[idx * 8];
        const int row = idx >> 3;
        const int u   = (idx & 7) * 8;
        *(v8h*)(Vtp + (size_t)row * S_LEN + n0 + u) = *(const v8h*)&tV[row * BN + u];
    }
}

// ---------------------------------------------------------------------------
// Main: transposed flash attention, f16-pre-staged K/V. Per wave: S^T = K (Q*s)^T
// + bias^T and O^T = V^T P^T; C-layout N axis = query -> per-lane scalar softmax.
// ---------------------------------------------------------------------------
__global__ __launch_bounds__(NTHREADS)
void fa_fwd_f16_kernel(const float* __restrict__ Q, const _Float16* __restrict__ Kh,
                       const _Float16* __restrict__ Vt, const float* __restrict__ Bias,
                       float* __restrict__ O)
{
    __shared__ _Float16 sK[BN * DIM];        // [key][d]      (8 KB)
    __shared__ _Float16 sV[DIM * BN];        // [d][key]      (8 KB)
    __shared__ _Float16 sP[NWAVE][16 * BN];  // [q][key] P^T exchange (16 KB)

    const int tid  = threadIdx.x;
    const int lane = tid & 31;
    const int wave = tid >> 5;
    const int h    = lane >> 4;
    const int ln   = lane & 15;

    const int q0 = blockIdx.x * BM;
    const int bh = blockIdx.y;

    const float*    Qp  = Q  + (size_t)bh * S_LEN * DIM;
    const _Float16* Khp = Kh + (size_t)bh * S_LEN * DIM;
    const _Float16* Vtp = Vt + (size_t)bh * DIM * S_LEN;
    float*          Op  = O  + (size_t)bh * S_LEN * DIM;

    const int qg = q0 + wave * 16 + ln;

    // Q^T B-fragments (loaded once; k order chosen here, same as A: element e -> K = e + 16*h)
    v16h qb0, qb1;
    {
        const float* qrow = Qp + (size_t)qg * DIM;
#pragma unroll
        for (int e = 0; e < 16; ++e) {
            qb0[e] = (_Float16)(qrow[e + 16 * h]      * 0.125f);
            qb1[e] = (_Float16)(qrow[e + 16 * h + 32] * 0.125f);
        }
    }

    float m_run = -1e30f, l_run = 0.f;
    const v8f vzero = {0.f,0.f,0.f,0.f,0.f,0.f,0.f,0.f};
    v8f oacc[4];
#pragma unroll
    for (int d = 0; d < 4; ++d) oacc[d] = vzero;

    for (int t = 0; t < S_LEN / BN; ++t) {
        const int n0 = t * BN;

        __syncthreads();

        // ---- Lean staging: pure f16 b128 copies (no cvt, no scatter)
        const _Float16* ksrc = Khp + (size_t)n0 * DIM;   // contiguous 64x64 tile
#pragma unroll
        for (int i = 0; i < 2; ++i) {
            const int idx = tid + i * NTHREADS;          // 512 units of 8 halves
            *(v8h*)&sK[idx * 8] = *(const v8h*)(ksrc + (size_t)idx * 8);
            const int row = idx >> 3;
            const int u   = (idx & 7) * 8;
            *(v8h*)&sV[row * BN + u] = *(const v8h*)(Vtp + (size_t)row * S_LEN + n0 + u);
        }
        __syncthreads();

        // ---- S^T subtiles
        float s[4][8];
        const float* brow = Bias + (size_t)qg * S_LEN + n0 + 8 * h;
#pragma unroll
        for (int ks = 0; ks < 4; ++ks) {
            v16h ka0, ka1;
            const _Float16* krow = &sK[(ks * 16 + ln) * DIM];
#pragma unroll
            for (int e = 0; e < 16; ++e) {               // 8-half runs -> ds_load_b128
                const int d = e + 16 * h;
                ka0[e] = krow[d];
                ka1[e] = krow[d + 32];
            }
            v8f c = vzero;
            c = __builtin_amdgcn_wmma_f32_16x16x32_f16(false, ka0, false, qb0, (short)0, c, false, false);
            c = __builtin_amdgcn_wmma_f32_16x16x32_f16(false, ka1, false, qb1, (short)0, c, false, false);
#pragma unroll
            for (int r = 0; r < 8; ++r)
                s[ks][r] = c[r] + brow[ks * 16 + r];
        }

        // ---- Online softmax: in-lane tree + one half-wave swap
        float mx = -1e30f;
#pragma unroll
        for (int ks = 0; ks < 4; ++ks)
#pragma unroll
            for (int r = 0; r < 8; ++r) mx = fmaxf(mx, s[ks][r]);
        mx = fmaxf(mx, __shfl_xor(mx, 16));
        const float mnew  = fmaxf(m_run, mx);
        const float alpha = __expf(m_run - mnew);
        m_run = mnew;
        float rs = 0.f;
#pragma unroll
        for (int ks = 0; ks < 4; ++ks)
#pragma unroll
            for (int r = 0; r < 8; ++r) {
                const float p = __expf(s[ks][r] - mnew);
                s[ks][r] = p;
                rs += p;
            }
        rs += __shfl_xor(rs, 16);
        l_run = l_run * alpha + rs;
#pragma unroll
        for (int d = 0; d < 4; ++d) oacc[d] = oacc[d] * alpha;

        // ---- P^T exchange via per-wave LDS (wave-internal; LDS in-order per wave)
        _Float16* pw = sP[wave];
#pragma unroll
        for (int ks = 0; ks < 4; ++ks) {
            v8h pk;
#pragma unroll
            for (int r = 0; r < 8; ++r) pk[r] = (_Float16)s[ks][r];
            *(v8h*)&pw[ln * BN + ks * 16 + 8 * h] = pk;
        }
        asm volatile("" ::: "memory");
        const v16h pb0 = *(const v16h*)&pw[ln * BN + 16 * h];
        const v16h pb1 = *(const v16h*)&pw[ln * BN + 32 + 16 * h];

        // ---- O^T += V^T P^T
#pragma unroll
        for (int ds = 0; ds < 4; ++ds) {
            v16h va0, va1;
            const _Float16* vrow = &sV[(ds * 16 + ln) * BN];
#pragma unroll
            for (int e = 0; e < 16; ++e) {
                const int k = e + 16 * h;
                va0[e] = vrow[k];
                va1[e] = vrow[k + 32];
            }
            oacc[ds] = __builtin_amdgcn_wmma_f32_16x16x32_f16(false, va0, false, pb0, (short)0, oacc[ds], false, false);
            oacc[ds] = __builtin_amdgcn_wmma_f32_16x16x32_f16(false, va1, false, pb1, (short)0, oacc[ds], false, false);
        }
    }

    {
        const float inv = 1.0f / l_run;
        float* orow = Op + (size_t)qg * DIM;
#pragma unroll
        for (int ds = 0; ds < 4; ++ds)
#pragma unroll
            for (int r = 0; r < 8; ++r)
                orow[ds * 16 + 8 * h + r] = oacc[ds][r] * inv;
    }
}

// ---------------------------------------------------------------------------
// Fallback (no workspace): round-2 kernel with in-loop f32->f16 staging.
// ---------------------------------------------------------------------------
__global__ __launch_bounds__(NTHREADS)
void fa_fwd_f32_kernel(const float* __restrict__ Q, const float* __restrict__ Km,
                       const float* __restrict__ Vm, const float* __restrict__ Bias,
                       float* __restrict__ O)
{
    __shared__ _Float16 sK[BN * DIM];
    __shared__ _Float16 sV[DIM * BN];
    __shared__ _Float16 sP[NWAVE][16 * BN];

    const int tid  = threadIdx.x;
    const int lane = tid & 31;
    const int wave = tid >> 5;
    const int h    = lane >> 4;
    const int ln   = lane & 15;

    const int q0 = blockIdx.x * BM;
    const int bh = blockIdx.y;

    const float* Qp = Q  + (size_t)bh * S_LEN * DIM;
    const float* Kp = Km + (size_t)bh * S_LEN * DIM;
    const float* Vp = Vm + (size_t)bh * S_LEN * DIM;
    float*       Op = O  + (size_t)bh * S_LEN * DIM;

    const int qg = q0 + wave * 16 + ln;

    v16h qb0, qb1;
    {
        const float* qrow = Qp + (size_t)qg * DIM;
#pragma unroll
        for (int e = 0; e < 16; ++e) {
            qb0[e] = (_Float16)(qrow[e + 16 * h]      * 0.125f);
            qb1[e] = (_Float16)(qrow[e + 16 * h + 32] * 0.125f);
        }
    }

    float m_run = -1e30f, l_run = 0.f;
    const v8f vzero = {0.f,0.f,0.f,0.f,0.f,0.f,0.f,0.f};
    v8f oacc[4];
#pragma unroll
    for (int d = 0; d < 4; ++d) oacc[d] = vzero;

    for (int t = 0; t < S_LEN / BN; ++t) {
        const int n0 = t * BN;
        __syncthreads();
#pragma unroll
        for (int i = 0; i < 4; ++i) {
            const int id4 = tid + i * NTHREADS;
            const int key = id4 >> 4;
            const int d4  = (id4 & 15) << 2;
            const float4 kf = *(const float4*)(Kp + (size_t)(n0 + key) * DIM + d4);
            _Float16* kd = &sK[key * DIM + d4];
            kd[0] = (_Float16)kf.x; kd[1] = (_Float16)kf.y;
            kd[2] = (_Float16)kf.z; kd[3] = (_Float16)kf.w;
            const float4 vf = *(const float4*)(Vp + (size_t)(n0 + key) * DIM + d4);
            sV[(d4 + 0) * BN + key] = (_Float16)vf.x;
            sV[(d4 + 1) * BN + key] = (_Float16)vf.y;
            sV[(d4 + 2) * BN + key] = (_Float16)vf.z;
            sV[(d4 + 3) * BN + key] = (_Float16)vf.w;
        }
        __syncthreads();

        float s[4][8];
        const float* brow = Bias + (size_t)qg * S_LEN + n0 + 8 * h;
#pragma unroll
        for (int ks = 0; ks < 4; ++ks) {
            v16h ka0, ka1;
            const _Float16* krow = &sK[(ks * 16 + ln) * DIM];
#pragma unroll
            for (int e = 0; e < 16; ++e) {
                const int d = e + 16 * h;
                ka0[e] = krow[d];
                ka1[e] = krow[d + 32];
            }
            v8f c = vzero;
            c = __builtin_amdgcn_wmma_f32_16x16x32_f16(false, ka0, false, qb0, (short)0, c, false, false);
            c = __builtin_amdgcn_wmma_f32_16x16x32_f16(false, ka1, false, qb1, (short)0, c, false, false);
#pragma unroll
            for (int r = 0; r < 8; ++r)
                s[ks][r] = c[r] + brow[ks * 16 + r];
        }

        float mx = -1e30f;
#pragma unroll
        for (int ks = 0; ks < 4; ++ks)
#pragma unroll
            for (int r = 0; r < 8; ++r) mx = fmaxf(mx, s[ks][r]);
        mx = fmaxf(mx, __shfl_xor(mx, 16));
        const float mnew  = fmaxf(m_run, mx);
        const float alpha = __expf(m_run - mnew);
        m_run = mnew;
        float rs = 0.f;
#pragma unroll
        for (int ks = 0; ks < 4; ++ks)
#pragma unroll
            for (int r = 0; r < 8; ++r) {
                const float p = __expf(s[ks][r] - mnew);
                s[ks][r] = p;
                rs += p;
            }
        rs += __shfl_xor(rs, 16);
        l_run = l_run * alpha + rs;
#pragma unroll
        for (int d = 0; d < 4; ++d) oacc[d] = oacc[d] * alpha;

        _Float16* pw = sP[wave];
#pragma unroll
        for (int ks = 0; ks < 4; ++ks) {
            v8h pk;
#pragma unroll
            for (int r = 0; r < 8; ++r) pk[r] = (_Float16)s[ks][r];
            *(v8h*)&pw[ln * BN + ks * 16 + 8 * h] = pk;
        }
        asm volatile("" ::: "memory");
        const v16h pb0 = *(const v16h*)&pw[ln * BN + 16 * h];
        const v16h pb1 = *(const v16h*)&pw[ln * BN + 32 + 16 * h];

#pragma unroll
        for (int ds = 0; ds < 4; ++ds) {
            v16h va0, va1;
            const _Float16* vrow = &sV[(ds * 16 + ln) * BN];
#pragma unroll
            for (int e = 0; e < 16; ++e) {
                const int k = e + 16 * h;
                va0[e] = vrow[k];
                va1[e] = vrow[k + 32];
            }
            oacc[ds] = __builtin_amdgcn_wmma_f32_16x16x32_f16(false, va0, false, pb0, (short)0, oacc[ds], false, false);
            oacc[ds] = __builtin_amdgcn_wmma_f32_16x16x32_f16(false, va1, false, pb1, (short)0, oacc[ds], false, false);
        }
    }

    {
        const float inv = 1.0f / l_run;
        float* orow = Op + (size_t)qg * DIM;
#pragma unroll
        for (int ds = 0; ds < 4; ++ds)
#pragma unroll
            for (int r = 0; r < 8; ++r)
                orow[ds * 16 + 8 * h + r] = oacc[ds][r] * inv;
    }
}

extern "C" void kernel_launch(void* const* d_in, const int* in_sizes, int n_in,
                              void* d_out, int out_size, void* d_ws, size_t ws_size,
                              hipStream_t stream) {
    const float* Q    = (const float*)d_in[0];   // mat1 [B,H,S,D]
    const float* K    = (const float*)d_in[1];   // mat2 [B,H,S,D]
    const float* V    = (const float*)d_in[2];   // mat3 [B,H,S,D]
    const float* Bias = (const float*)d_in[3];   // bias [1,1,S,S]
    float* Out        = (float*)d_out;

    const int BH = in_sizes[0] / (S_LEN * DIM);  // B*H = 32
    const size_t elems = (size_t)BH * S_LEN * DIM;
    const size_t need  = 2 * elems * sizeof(_Float16);  // Kh + Vt

    dim3 block(NTHREADS);
    if (d_ws != nullptr && ws_size >= need) {
        _Float16* Kh = (_Float16*)d_ws;
        _Float16* Vt = Kh + elems;
        dim3 gridPre(S_LEN / BN, BH);
        preconvert_kernel<<<gridPre, block, 0, stream>>>(K, V, Kh, Vt);
        dim3 gridFa(S_LEN / BM, BH);
        fa_fwd_f16_kernel<<<gridFa, block, 0, stream>>>(Q, Kh, Vt, Bias, Out);
    } else {
        dim3 gridFa(S_LEN / BM, BH);
        fa_fwd_f32_kernel<<<gridFa, block, 0, stream>>>(Q, K, V, Bias, Out);
    }
}
